// Net_38087769981655
// MI455X (gfx1250) — matrix operand order repaired and checked by reading, not compiled, not run on hardware
//
#include <hip/hip_runtime.h>
#include <math.h>

#define Nn    100000
#define Ee    3200000
#define F_IN  512
#define H1c   8
#define D1c   8
#define F1    (H1c * D1c)   // 64
#define Cc    16
#define NEG_SLOPE 0.2f

typedef __attribute__((ext_vector_type(16))) __bf16 v16bf;
typedef __attribute__((ext_vector_type(8)))  float  v8f;

// ---------------------------------------------------------------------------
// WMMA GEMM: C[M x NC] = A[M x K] @ B[K x NC]   (fp32 in/out, bf16-split WMMA)
// One wave (32 threads) per 16x16 output tile. K % 32 == 0, NC % 16 == 0.
// ---------------------------------------------------------------------------
__global__ __launch_bounds__(32)
void wmma_gemm_kernel(const float* __restrict__ A, const float* __restrict__ B,
                      float* __restrict__ C, int M, int K, int NC) {
    __shared__ float As[16][33];
    __shared__ float Bs[32][17];
    const int lane = threadIdx.x;          // 0..31
    const int mt = blockIdx.x * 16;
    const int nt = blockIdx.y * 16;
    const int hi = (lane >= 16) ? 1 : 0;
    const int m16 = lane & 15;

    v8f acc = {};
    for (int k0 = 0; k0 < K; k0 += 32) {
        // stage A tile 16x32 (rows of nodes), zero-pad past M
        for (int i = lane; i < 16 * 32; i += 32) {
            int r = i >> 5, c = i & 31;
            int row = mt + r;
            As[r][c] = (row < M) ? A[(long)row * K + (k0 + c)] : 0.0f;
        }
        // stage B tile 32x16
        for (int i = lane; i < 32 * 16; i += 32) {
            int r = i >> 4, c = i & 15;
            Bs[r][c] = B[(long)(k0 + r) * NC + (nt + c)];
        }
        __syncthreads();

        // Build fragments per ISA 7.12.2 (16-bit A 16x32, B 32x16), with
        // hi/lo bf16 split of the fp32 data for near-fp32 accuracy.
        v16bf ah, al, bh, bl;
#pragma unroll
        for (int e = 0; e < 16; ++e) {
            int v = e >> 1, half = e & 1;
            int ka = ((v < 4) ? (2 * v) : (16 + 2 * (v - 4))) + half + (hi ? 8 : 0);
            float fa = As[m16][ka];
            __bf16 fah = (__bf16)fa;
            ah[e] = fah;
            al[e] = (__bf16)(fa - (float)fah);

            int kb = ((e < 8) ? e : (16 + (e - 8))) + (hi ? 8 : 0);
            float fb = Bs[kb][m16];
            __bf16 fbh = (__bf16)fb;
            bh[e] = fbh;
            bl[e] = (__bf16)(fb - (float)fbh);
        }
        acc = __builtin_amdgcn_wmma_f32_16x16x32_bf16(false, ah, false, bh, (short)0, acc, false, false);
        acc = __builtin_amdgcn_wmma_f32_16x16x32_bf16(false, ah, false, bl, (short)0, acc, false, false);
        acc = __builtin_amdgcn_wmma_f32_16x16x32_bf16(false, al, false, bh, (short)0, acc, false, false);
        __syncthreads();
    }

    // C/D layout: VGPR r -> row r (lanes 0-15) / row r+8 (lanes 16-31), col = lane%16
#pragma unroll
    for (int r = 0; r < 8; ++r) {
        int row = mt + r + (hi ? 8 : 0);
        int col = nt + m16;
        if (row < M) C[(long)row * NC + col] = acc[r];
    }
}

// ---------------------------------------------------------------------------
// Elementwise / helper kernels
// ---------------------------------------------------------------------------
__global__ void fill_kernel(float* __restrict__ p, int n, float v) {
    int i = blockIdx.x * blockDim.x + threadIdx.x;
    if (i < n) p[i] = v;
}

__global__ void elu_kernel(float* __restrict__ p, int n) {
    int i = blockIdx.x * blockDim.x + threadIdx.x;
    if (i < n) {
        float v = p[i];
        p[i] = (v > 0.0f) ? v : expm1f(v);
    }
}

// attention logits: als[n,h] = sum_d h[n,h,d]*a_src[h,d]; same for ald
__global__ void att_logits_kernel(const float* __restrict__ h,
                                  const float* __restrict__ a_src,
                                  const float* __restrict__ a_dst,
                                  float* __restrict__ als, float* __restrict__ ald,
                                  int n, int H, int D) {
    int i = blockIdx.x * blockDim.x + threadIdx.x;
    if (i >= n * H) return;
    int node = i / H, hh = i % H;
    const float* hp = h + (long)node * H * D + hh * D;
    float s = 0.0f, d = 0.0f;
    for (int k = 0; k < D; ++k) {
        s += hp[k] * a_src[hh * D + k];
        d += hp[k] * a_dst[hh * D + k];
    }
    als[i] = s;
    ald[i] = d;
}

__device__ __forceinline__ float lrelu(float v) {
    return v > 0.0f ? v : NEG_SLOPE * v;
}

__device__ __forceinline__ void atomic_max_float(float* addr, float val) {
    unsigned int* ua = (unsigned int*)addr;
    unsigned int old = __float_as_uint(*addr);
    while (__uint_as_float(old) < val) {
        unsigned int assumed = old;
        old = atomicCAS(ua, assumed, __float_as_uint(val));
        if (old == assumed) break;
    }
}

// pass 1: segment max of leaky-relu logits keyed by dst
__global__ void edge_max_kernel(const int* __restrict__ src, const int* __restrict__ dst,
                                const float* __restrict__ als, const float* __restrict__ ald,
                                float* __restrict__ emax, int E, int H) {
    int i = blockIdx.x * blockDim.x + threadIdx.x;
    if (i >= E * H) return;
    int e = i / H, hh = i % H;
    int s = src[e], d = dst[e];
    float v = lrelu(als[s * H + hh] + ald[d * H + hh]);
    atomic_max_float(&emax[d * H + hh], v);
}

// pass 2: segment sum of exp(e - max)
__global__ void edge_sum_kernel(const int* __restrict__ src, const int* __restrict__ dst,
                                const float* __restrict__ als, const float* __restrict__ ald,
                                const float* __restrict__ emax, float* __restrict__ esum,
                                int E, int H) {
    int i = blockIdx.x * blockDim.x + threadIdx.x;
    if (i >= E * H) return;
    int e = i / H, hh = i % H;
    int s = src[e], d = dst[e];
    float v = lrelu(als[s * H + hh] + ald[d * H + hh]);
    atomicAdd(&esum[d * H + hh], expf(v - emax[d * H + hh]));
}

// pass 3: agg[dst,h,:] += h[src,h,:] * alpha
__global__ void edge_agg_kernel(const int* __restrict__ src, const int* __restrict__ dst,
                                const float* __restrict__ als, const float* __restrict__ ald,
                                const float* __restrict__ emax, const float* __restrict__ esum,
                                const float* __restrict__ h, float* __restrict__ agg,
                                int E, int H, int D) {
    int i = blockIdx.x * blockDim.x + threadIdx.x;
    if (i >= E * H) return;
    int e = i / H, hh = i % H;
    int s = src[e], d = dst[e];
    float v = lrelu(als[s * H + hh] + ald[d * H + hh]);
    float ex = expf(v - emax[d * H + hh]);
    float alpha = ex / (esum[d * H + hh] + 1e-16f);
    const float* hp = h + (long)s * H * D + hh * D;
    float* op = agg + (long)d * H * D + hh * D;
    for (int k = 0; k < D; ++k) atomicAdd(&op[k], hp[k] * alpha);
}

__global__ void logsoftmax_kernel(const float* __restrict__ in, float* __restrict__ out, int n) {
    int i = blockIdx.x * blockDim.x + threadIdx.x;
    if (i >= n) return;
    const float* p = in + (long)i * Cc;
    float m = p[0];
    for (int c = 1; c < Cc; ++c) m = fmaxf(m, p[c]);
    float s = 0.0f;
    for (int c = 0; c < Cc; ++c) s += expf(p[c] - m);
    float l = logf(s);
    for (int c = 0; c < Cc; ++c) out[(long)i * Cc + c] = p[c] - m - l;
}

// ---------------------------------------------------------------------------
// Host launcher
// ---------------------------------------------------------------------------
extern "C" void kernel_launch(void* const* d_in, const int* in_sizes, int n_in,
                              void* d_out, int out_size, void* d_ws, size_t ws_size,
                              hipStream_t stream) {
    const float* x    = (const float*)d_in[0];
    const int*   ei   = (const int*)  d_in[1];
    const float* W1   = (const float*)d_in[2];
    const float* a1s  = (const float*)d_in[3];
    const float* a1d  = (const float*)d_in[4];
    const float* W2   = (const float*)d_in[5];
    const float* a2s  = (const float*)d_in[6];
    const float* a2d  = (const float*)d_in[7];
    float* out = (float*)d_out;

    const int E = in_sizes[1] / 2;
    const int* src = ei;
    const int* dst = ei + E;

    // workspace partition (floats)
    float* w = (float*)d_ws;
    float* h1raw = w;                 w += (size_t)Nn * F1;  // 6.4M
    float* al1s  = w;                 w += (size_t)Nn * H1c;
    float* al1d  = w;                 w += (size_t)Nn * H1c;
    float* emax1 = w;                 w += (size_t)Nn * H1c;
    float* esum1 = w;                 w += (size_t)Nn * H1c;
    float* agg1  = w;                 w += (size_t)Nn * F1;  // becomes h1 after ELU
    float* h2raw = w;                 w += (size_t)Nn * Cc;
    float* al2s  = w;                 w += (size_t)Nn;
    float* al2d  = w;                 w += (size_t)Nn;
    float* emax2 = w;                 w += (size_t)Nn;
    float* esum2 = w;                 w += (size_t)Nn;
    float* agg2  = w;                 w += (size_t)Nn * Cc;

    const int TB = 256;
    auto blocks = [](long n, int tb) { return (unsigned)((n + tb - 1) / tb); };
    const int mtiles = (Nn + 15) / 16;

    // ---------------- layer 1 ----------------
    wmma_gemm_kernel<<<dim3(mtiles, F1 / 16), 32, 0, stream>>>(x, W1, h1raw, Nn, F_IN, F1);
    att_logits_kernel<<<blocks((long)Nn * H1c, TB), TB, 0, stream>>>(h1raw, a1s, a1d, al1s, al1d, Nn, H1c, D1c);
    fill_kernel<<<blocks((long)Nn * H1c, TB), TB, 0, stream>>>(emax1, Nn * H1c, -INFINITY);
    fill_kernel<<<blocks((long)Nn * H1c, TB), TB, 0, stream>>>(esum1, Nn * H1c, 0.0f);
    fill_kernel<<<blocks((long)Nn * F1, TB), TB, 0, stream>>>(agg1, Nn * F1, 0.0f);
    edge_max_kernel<<<blocks((long)E * H1c, TB), TB, 0, stream>>>(src, dst, al1s, al1d, emax1, E, H1c);
    edge_sum_kernel<<<blocks((long)E * H1c, TB), TB, 0, stream>>>(src, dst, al1s, al1d, emax1, esum1, E, H1c);
    edge_agg_kernel<<<blocks((long)E * H1c, TB), TB, 0, stream>>>(src, dst, al1s, al1d, emax1, esum1,
                                                                  h1raw, agg1, E, H1c, D1c);
    elu_kernel<<<blocks((long)Nn * F1, TB), TB, 0, stream>>>(agg1, Nn * F1);

    // ---------------- layer 2 ----------------
    wmma_gemm_kernel<<<dim3(mtiles, Cc / 16), 32, 0, stream>>>(agg1, W2, h2raw, Nn, F1, Cc);
    att_logits_kernel<<<blocks((long)Nn, TB), TB, 0, stream>>>(h2raw, a2s, a2d, al2s, al2d, Nn, 1, Cc);
    fill_kernel<<<blocks((long)Nn, TB), TB, 0, stream>>>(emax2, Nn, -INFINITY);
    fill_kernel<<<blocks((long)Nn, TB), TB, 0, stream>>>(esum2, Nn, 0.0f);
    fill_kernel<<<blocks((long)Nn * Cc, TB), TB, 0, stream>>>(agg2, Nn * Cc, 0.0f);
    edge_max_kernel<<<blocks((long)E, TB), TB, 0, stream>>>(src, dst, al2s, al2d, emax2, E, 1);
    edge_sum_kernel<<<blocks((long)E, TB), TB, 0, stream>>>(src, dst, al2s, al2d, emax2, esum2, E, 1);
    edge_agg_kernel<<<blocks((long)E, TB), TB, 0, stream>>>(src, dst, al2s, al2d, emax2, esum2,
                                                            h2raw, agg2, E, 1, Cc);
    logsoftmax_kernel<<<blocks((long)Nn, TB), TB, 0, stream>>>(agg2, out, Nn);
}
